// PointNetSetAbstractionMsg_46969762349058
// MI455X (gfx1250) — matrix operand order repaired and checked by reading, not compiled, not run on hardware
//
#include <hip/hip_runtime.h>
#include <hip/hip_bf16.h>
#include <math.h>

#define B_      8
#define N_      4096
#define S_      1024
#define IN_CH_  64

typedef __attribute__((ext_vector_type(16))) __bf16 v16bf;
typedef __attribute__((ext_vector_type(8)))  float  v8f;

__device__ __forceinline__ unsigned short f2bf(float f) {
    unsigned int x = __float_as_uint(f);
    x += 0x7FFFu + ((x >> 16) & 1u);            // round-to-nearest-even
    return (unsigned short)(x >> 16);
}
__device__ __forceinline__ float bf2f(unsigned short h) {
    return __uint_as_float(((unsigned int)h) << 16);
}

// ---------------------------------------------------------------- FPS
__global__ __launch_bounds__(1024)
void fps_kernel(const float* __restrict__ xyz, int* __restrict__ fps_idx) {
    const int b = blockIdx.x;
    const int tid = threadIdx.x;
    __shared__ float dist[N_];
    __shared__ float rv[1024];
    __shared__ int   ri[1024];
    __shared__ int   sFar;
    __shared__ float c3[3];

    for (int n = tid; n < N_; n += 1024) dist[n] = 1e10f;
    if (tid == 0) sFar = 0;
    __syncthreads();

    const float* xb = xyz + (size_t)b * N_ * 3;
    for (int it = 0; it < 1024; ++it) {
        if (tid == 0) {
            int f = sFar;
            fps_idx[b * 1024 + it] = f;
            c3[0] = xb[f * 3 + 0]; c3[1] = xb[f * 3 + 1]; c3[2] = xb[f * 3 + 2];
        }
        __syncthreads();
        const float cx = c3[0], cy = c3[1], cz = c3[2];
        float bv = -1.0f; int bi = 0;
        for (int n = tid; n < N_; n += 1024) {
            float dx = xb[n * 3 + 0] - cx;
            float dy = xb[n * 3 + 1] - cy;
            float dz = xb[n * 3 + 2] - cz;
            float d  = dx * dx + dy * dy + dz * dz;
            float nd = fminf(dist[n], d);
            dist[n] = nd;
            if (nd > bv) { bv = nd; bi = n; }
        }
        rv[tid] = bv; ri[tid] = bi;
        __syncthreads();
        for (int s = 512; s > 0; s >>= 1) {
            if (tid < s) {
                float v2 = rv[tid + s]; int i2 = ri[tid + s];
                if (v2 > rv[tid] || (v2 == rv[tid] && i2 < ri[tid])) { rv[tid] = v2; ri[tid] = i2; }
            }
            __syncthreads();
        }
        if (tid == 0) sFar = ri[0];
        __syncthreads();
    }
}

// ---------------------------------------------------------------- new_xyz gather
__global__ void newxyz_kernel(const float* __restrict__ xyz,
                              const int* __restrict__ fps_idx,
                              float* __restrict__ out) {
    int t = blockIdx.x * 256 + threadIdx.x;
    if (t >= B_ * S_) return;
    int b = t / S_;
    int id = fps_idx[t];
    const float* p = xyz + ((size_t)b * N_ + id) * 3;
    out[t * 3 + 0] = p[0]; out[t * 3 + 1] = p[1]; out[t * 3 + 2] = p[2];
}

// ---------------------------------------------------------------- kNN: 128 smallest, sorted
__global__ __launch_bounds__(256)
void knn_kernel(const float* __restrict__ xyz, const float* __restrict__ newxyz,
                int* __restrict__ idx128, float* __restrict__ dist128) {
    const int bs = blockIdx.x;            // b*S + s
    const int b  = bs / S_;
    const int tid = threadIdx.x;
    __shared__ float d[N_];
    __shared__ float rv[256];
    __shared__ int   ri[256];

    const float cx = newxyz[bs * 3 + 0], cy = newxyz[bs * 3 + 1], cz = newxyz[bs * 3 + 2];
    const float* xb = xyz + (size_t)b * N_ * 3;
    for (int n = tid; n < N_; n += 256) {
        float dx = xb[n * 3 + 0] - cx, dy = xb[n * 3 + 1] - cy, dz = xb[n * 3 + 2] - cz;
        d[n] = dx * dx + dy * dy + dz * dz;
    }
    __syncthreads();
    for (int j = 0; j < 128; ++j) {
        float bv = 1e30f; int bi = N_ - 1;
        for (int n = tid; n < N_; n += 256) {
            float v = d[n];
            if (v < bv) { bv = v; bi = n; }
        }
        rv[tid] = bv; ri[tid] = bi;
        __syncthreads();
        for (int s = 128; s > 0; s >>= 1) {
            if (tid < s) {
                float v2 = rv[tid + s]; int i2 = ri[tid + s];
                if (v2 < rv[tid] || (v2 == rv[tid] && i2 < ri[tid])) { rv[tid] = v2; ri[tid] = i2; }
            }
            __syncthreads();
        }
        if (tid == 0) {
            idx128[(size_t)bs * 128 + j]  = ri[0];
            dist128[(size_t)bs * 128 + j] = rv[0];
            d[ri[0]] = 1e30f;
        }
        __syncthreads();
    }
}

// ---------------------------------------------------------------- stats helpers
__global__ void zero_stats_kernel(float* sum, float* sumsq) {
    int t = threadIdx.x;
    if (t < 128) { sum[t] = 0.f; sumsq[t] = 0.f; }
}

__global__ void bn_prepare_kernel(const float* __restrict__ sum, const float* __restrict__ sumsq,
                                  const float* __restrict__ gamma, const float* __restrict__ beta,
                                  float* __restrict__ a, float* __restrict__ shift,
                                  int C, float invCnt) {
    int c = threadIdx.x;
    if (c < C) {
        float mean = sum[c] * invCnt;
        float var  = sumsq[c] * invCnt - mean * mean;
        float s    = gamma[c] * rsqrtf(var + 1e-5f);
        a[c] = s;
        shift[c] = beta[c] - mean * s;
    }
}

// ---------------------------------------------------------------- WMMA GEMM (bf16 x bf16 -> f32)
// MODE 0: A = gather/concat(xyz_local, points) with radius mask (layer 1)
// MODE 1: A = relu(featIn * bnA + bnB) from bf16 activations (layers 2/3)
template <int MODE, int CIN, int KPAD, int COUT>
__global__ __launch_bounds__(256)
void mlp_gemm_kernel(int ns, float r2,
                     const float* __restrict__ xyz, const float* __restrict__ points,
                     const float* __restrict__ newxyz,
                     const int* __restrict__ idx128, const float* __restrict__ dist128,
                     const unsigned short* __restrict__ featIn,
                     const float* __restrict__ bnA, const float* __restrict__ bnB,
                     const float* __restrict__ W, const float* __restrict__ bias,
                     unsigned short* __restrict__ featOut,
                     float* __restrict__ statSum, float* __restrict__ statSumsq) {
    constexpr int NT = COUT / 16;                       // 16-col tiles per wave
    __shared__ __align__(16) unsigned short sA[128 * KPAD];  // A tile, row-major [row][k]
    __shared__ __align__(16) unsigned short sW[COUT * KPAD]; // W transposed [o][k]
    __shared__ float sBias[COUT];
    __shared__ float sBnA[KPAD], sBnB[KPAD];
    __shared__ int   sIdx[128], sB[128], sS[128];
    __shared__ float sSum[COUT], sSumsq[COUT];

    const int tid = threadIdx.x;
    const long long row0 = (long long)blockIdx.x * 128;

    // weights -> LDS, transposed, bf16, zero-padded to KPAD
    for (int e = tid; e < COUT * KPAD; e += 256) {
        int o = e / KPAD, k = e % KPAD;
        float w = (k < CIN) ? W[(size_t)k * COUT + o] : 0.f;
        sW[o * KPAD + k] = f2bf(w);
    }
    for (int c = tid; c < COUT; c += 256) { sBias[c] = bias[c]; sSum[c] = 0.f; sSumsq[c] = 0.f; }
    if (MODE == 1) {
        for (int c = tid; c < CIN; c += 256) { sBnA[c] = bnA[c]; sBnB[c] = bnB[c]; }
    } else if (tid < 128) {
        long long r = row0 + tid;
        int b   = (int)(r / ((long long)S_ * ns));
        int rem = (int)(r % ((long long)S_ * ns));
        int s = rem / ns, k = rem % ns;
        long long bs = (long long)b * S_ + s;
        int   raw = idx128[bs * 128 + k];
        float dv  = dist128[bs * 128 + k];
        sIdx[tid] = (dv > r2) ? (N_ - 1) : raw;
        sB[tid] = b; sS[tid] = s;
    }
    __syncthreads();

    // A tile -> LDS bf16
    if (MODE == 0) {
        for (int e = tid; e < 128 * KPAD; e += 256) {
            int row = e / KPAD, c = e % KPAD;
            float v = 0.f;
            if (c < CIN) {
                int b = sB[row], s = sS[row], id = sIdx[row];
                if (c < 3) v = xyz[((size_t)b * N_ + id) * 3 + c] - newxyz[((size_t)b * S_ + s) * 3 + c];
                else       v = points[((size_t)b * N_ + id) * IN_CH_ + (c - 3)];
            }
            sA[row * KPAD + c] = f2bf(v);
        }
    } else {
        for (int e = tid; e < 128 * KPAD; e += 256) {
            int row = e / KPAD, c = e % KPAD;
            float v = 0.f;
            if (c < CIN) {
                float x = bf2f(featIn[(row0 + row) * CIN + c]);
                v = fmaxf(0.f, x * sBnA[c] + sBnB[c]);
            }
            sA[row * KPAD + c] = f2bf(v);
        }
    }
    __syncthreads();

    const int wave  = tid >> 5;
    const int lane  = tid & 31;
    const int m0    = wave * 16;                 // 8 waves x 16 rows = 128 rows/block
    const int half8 = (lane < 16) ? 0 : 8;
    const int arow  = m0 + (lane & 15);

    v8f acc[NT] = {};

#pragma unroll
    for (int kk = 0; kk < KPAD; kk += 32) {
        // A fragment: contiguous k = 16*(lane/16) + s, matching B
        union { v16bf v; unsigned int u[8]; } afrag;
#pragma unroll
        for (int j = 0; j < 8; ++j) {
            int kp = kk + 2 * half8 + 2 * j;
            afrag.u[j] = *(const unsigned int*)&sA[arow * KPAD + kp];
        }
#pragma unroll
        for (int t = 0; t < NT; ++t) {
            int col = t * 16 + (lane & 15);
            union { v16bf v; uint4 q[2]; } bfrag;
            const uint4* src = (const uint4*)&sW[col * KPAD + kk + ((lane < 16) ? 0 : 16)];
            bfrag.q[0] = src[0];
            bfrag.q[1] = src[1];
            acc[t] = __builtin_amdgcn_wmma_f32_16x16x32_bf16(
                false, afrag.v, false, bfrag.v, (short)0, acc[t], false, false);
        }
    }

    // epilogue: bias add, store bf16, per-channel partial sums
#pragma unroll
    for (int t = 0; t < NT; ++t) {
        int col = t * 16 + (lane & 15);
        float bcol = sBias[col];
        float psum = 0.f, psq = 0.f;
#pragma unroll
        for (int r = 0; r < 8; ++r) {
            float v = acc[t][r] + bcol;
            int rowl = m0 + r + ((lane < 16) ? 0 : 8);   // D-matrix VGPR->M mapping
            featOut[(row0 + rowl) * COUT + col] = f2bf(v);
            psum += v; psq += v * v;
        }
        atomicAdd(&sSum[col], psum);
        atomicAdd(&sSumsq[col], psq);
    }
    __syncthreads();
    for (int c = tid; c < COUT; c += 256) {
        atomicAdd(&statSum[c],   sSum[c]);
        atomicAdd(&statSumsq[c], sSumsq[c]);
    }
}

// ---------------------------------------------------------------- BN + ReLU + max over K
__global__ void finalize_kernel(const unsigned short* __restrict__ feat,
                                const float* __restrict__ bnA, const float* __restrict__ bnB,
                                int Cout, int ns, int chOff, float* __restrict__ out) {
    long long t = (long long)blockIdx.x * 256 + threadIdx.x;
    long long total = (long long)B_ * S_ * Cout;
    if (t >= total) return;
    int c = (int)(t % Cout);
    long long bs = t / Cout;
    float a = bnA[c], sh = bnB[c];
    float m = -1e30f;
    const unsigned short* f = feat + bs * (long long)ns * Cout + c;
    for (int k = 0; k < ns; ++k) {
        float v = fmaxf(0.f, bf2f(f[(long long)k * Cout]) * a + sh);
        m = fmaxf(m, v);
    }
    out[bs * 320 + chOff + c] = m;
}

// ---------------------------------------------------------------- driver
extern "C" void kernel_launch(void* const* d_in, const int* in_sizes, int n_in,
                              void* d_out, int out_size, void* d_ws, size_t ws_size,
                              hipStream_t stream) {
    (void)in_sizes; (void)n_in; (void)out_size; (void)ws_size;
    const float* xyz    = (const float*)d_in[0];
    const float* points = (const float*)d_in[1];
    // params flattened per JAX pytree: scale-major, layer-major, dict keys sorted: W, b, beta, gamma
    auto Wp = [&](int i, int j) { return (const float*)d_in[2 + (i * 3 + j) * 4 + 0]; };
    auto Bp = [&](int i, int j) { return (const float*)d_in[2 + (i * 3 + j) * 4 + 1]; };
    auto Be = [&](int i, int j) { return (const float*)d_in[2 + (i * 3 + j) * 4 + 2]; };
    auto Ga = [&](int i, int j) { return (const float*)d_in[2 + (i * 3 + j) * 4 + 3]; };

    char* ws = (char*)d_ws;
    int*   fps_idx  = (int*)ws;              ws += (size_t)B_ * S_ * sizeof(int);
    int*   idx128   = (int*)ws;              ws += (size_t)B_ * S_ * 128 * sizeof(int);
    float* dist128  = (float*)ws;            ws += (size_t)B_ * S_ * 128 * sizeof(float);
    float* statSum  = (float*)ws;            ws += 128 * sizeof(float);
    float* statSumsq= (float*)ws;            ws += 128 * sizeof(float);
    float* bnA      = (float*)ws;            ws += 128 * sizeof(float);
    float* bnB      = (float*)ws;            ws += 128 * sizeof(float);
    unsigned short* featA = (unsigned short*)ws; ws += (size_t)B_ * S_ * 128 * 128 * sizeof(unsigned short);
    unsigned short* featB = (unsigned short*)ws;

    float* newxyz = (float*)d_out;                 // (8,1024,3)
    float* newpts = (float*)d_out + B_ * S_ * 3;   // (8,1024,320)

    fps_kernel<<<B_, 1024, 0, stream>>>(xyz, fps_idx);
    newxyz_kernel<<<(B_ * S_ + 255) / 256, 256, 0, stream>>>(xyz, fps_idx, newxyz);
    knn_kernel<<<B_ * S_, 256, 0, stream>>>(xyz, newxyz, idx128, dist128);

    const unsigned short* nullu16 = nullptr;
    const float* nullf = nullptr;

    // -------- scale 0: ns=16, r=0.1, mlp 67->32->32->64
    {
        const int ns = 16; const float r2 = 0.1f * 0.1f;
        long long rows = (long long)B_ * S_ * ns;
        int blocks = (int)(rows / 128);
        float invCnt = 1.f / (float)rows;

        zero_stats_kernel<<<1, 128, 0, stream>>>(statSum, statSumsq);
        mlp_gemm_kernel<0, 67, 96, 32><<<blocks, 256, 0, stream>>>(ns, r2,
            xyz, points, newxyz, idx128, dist128, nullu16, nullf, nullf,
            Wp(0, 0), Bp(0, 0), featA, statSum, statSumsq);
        bn_prepare_kernel<<<1, 128, 0, stream>>>(statSum, statSumsq, Ga(0, 0), Be(0, 0), bnA, bnB, 32, invCnt);

        zero_stats_kernel<<<1, 128, 0, stream>>>(statSum, statSumsq);
        mlp_gemm_kernel<1, 32, 32, 32><<<blocks, 256, 0, stream>>>(ns, r2,
            xyz, points, newxyz, idx128, dist128, featA, bnA, bnB,
            Wp(0, 1), Bp(0, 1), featB, statSum, statSumsq);
        bn_prepare_kernel<<<1, 128, 0, stream>>>(statSum, statSumsq, Ga(0, 1), Be(0, 1), bnA, bnB, 32, invCnt);

        zero_stats_kernel<<<1, 128, 0, stream>>>(statSum, statSumsq);
        mlp_gemm_kernel<1, 32, 32, 64><<<blocks, 256, 0, stream>>>(ns, r2,
            xyz, points, newxyz, idx128, dist128, featB, bnA, bnB,
            Wp(0, 2), Bp(0, 2), featA, statSum, statSumsq);
        bn_prepare_kernel<<<1, 128, 0, stream>>>(statSum, statSumsq, Ga(0, 2), Be(0, 2), bnA, bnB, 64, invCnt);

        long long tot = (long long)B_ * S_ * 64;
        finalize_kernel<<<(int)((tot + 255) / 256), 256, 0, stream>>>(featA, bnA, bnB, 64, ns, 0, newpts);
    }

    // -------- scale 1: ns=32, r=0.2, mlp 67->64->64->128
    {
        const int ns = 32; const float r2 = 0.2f * 0.2f;
        long long rows = (long long)B_ * S_ * ns;
        int blocks = (int)(rows / 128);
        float invCnt = 1.f / (float)rows;

        zero_stats_kernel<<<1, 128, 0, stream>>>(statSum, statSumsq);
        mlp_gemm_kernel<0, 67, 96, 64><<<blocks, 256, 0, stream>>>(ns, r2,
            xyz, points, newxyz, idx128, dist128, nullu16, nullf, nullf,
            Wp(1, 0), Bp(1, 0), featA, statSum, statSumsq);
        bn_prepare_kernel<<<1, 128, 0, stream>>>(statSum, statSumsq, Ga(1, 0), Be(1, 0), bnA, bnB, 64, invCnt);

        zero_stats_kernel<<<1, 128, 0, stream>>>(statSum, statSumsq);
        mlp_gemm_kernel<1, 64, 64, 64><<<blocks, 256, 0, stream>>>(ns, r2,
            xyz, points, newxyz, idx128, dist128, featA, bnA, bnB,
            Wp(1, 1), Bp(1, 1), featB, statSum, statSumsq);
        bn_prepare_kernel<<<1, 128, 0, stream>>>(statSum, statSumsq, Ga(1, 1), Be(1, 1), bnA, bnB, 64, invCnt);

        zero_stats_kernel<<<1, 128, 0, stream>>>(statSum, statSumsq);
        mlp_gemm_kernel<1, 64, 64, 128><<<blocks, 256, 0, stream>>>(ns, r2,
            xyz, points, newxyz, idx128, dist128, featB, bnA, bnB,
            Wp(1, 2), Bp(1, 2), featA, statSum, statSumsq);
        bn_prepare_kernel<<<1, 128, 0, stream>>>(statSum, statSumsq, Ga(1, 2), Be(1, 2), bnA, bnB, 128, invCnt);

        long long tot = (long long)B_ * S_ * 128;
        finalize_kernel<<<(int)((tot + 255) / 256), 256, 0, stream>>>(featA, bnA, bnB, 128, ns, 64, newpts);
    }

    // -------- scale 2: ns=128, r=0.4, mlp 67->64->96->128
    {
        const int ns = 128; const float r2 = 0.4f * 0.4f;
        long long rows = (long long)B_ * S_ * ns;
        int blocks = (int)(rows / 128);
        float invCnt = 1.f / (float)rows;

        zero_stats_kernel<<<1, 128, 0, stream>>>(statSum, statSumsq);
        mlp_gemm_kernel<0, 67, 96, 64><<<blocks, 256, 0, stream>>>(ns, r2,
            xyz, points, newxyz, idx128, dist128, nullu16, nullf, nullf,
            Wp(2, 0), Bp(2, 0), featA, statSum, statSumsq);
        bn_prepare_kernel<<<1, 128, 0, stream>>>(statSum, statSumsq, Ga(2, 0), Be(2, 0), bnA, bnB, 64, invCnt);

        zero_stats_kernel<<<1, 128, 0, stream>>>(statSum, statSumsq);
        mlp_gemm_kernel<1, 64, 64, 96><<<blocks, 256, 0, stream>>>(ns, r2,
            xyz, points, newxyz, idx128, dist128, featA, bnA, bnB,
            Wp(2, 1), Bp(2, 1), featB, statSum, statSumsq);
        bn_prepare_kernel<<<1, 128, 0, stream>>>(statSum, statSumsq, Ga(2, 1), Be(2, 1), bnA, bnB, 96, invCnt);

        zero_stats_kernel<<<1, 128, 0, stream>>>(statSum, statSumsq);
        mlp_gemm_kernel<1, 96, 96, 128><<<blocks, 256, 0, stream>>>(ns, r2,
            xyz, points, newxyz, idx128, dist128, featB, bnA, bnB,
            Wp(2, 2), Bp(2, 2), featA, statSum, statSumsq);
        bn_prepare_kernel<<<1, 128, 0, stream>>>(statSum, statSumsq, Ga(2, 2), Be(2, 2), bnA, bnB, 128, invCnt);

        long long tot = (long long)B_ * S_ * 128;
        finalize_kernel<<<(int)((tot + 255) / 256), 256, 0, stream>>>(featA, bnA, bnB, 128, ns, 192, newpts);
    }
}
